// transformer_block_27015344292366
// MI455X (gfx1250) — matrix operand order repaired and checked by reading, not compiled, not run on hardware
//
#include <hip/hip_runtime.h>

typedef __bf16 bf16_t;
typedef __attribute__((ext_vector_type(16))) __bf16 v16bf;
typedef __attribute__((ext_vector_type(8)))  __bf16 bf16x8;
typedef __attribute__((ext_vector_type(8)))  float  v8f;

#define DEV static __device__ __forceinline__

// Problem constants
constexpr int Bb = 8192, Lx = 64, Cc = 128;
// ws layout (bytes)
constexpr size_t WS_WQKV  = 0;        // 128x384 bf16 = 98304
constexpr size_t WS_WPROJ = 98304;    // 128x128 bf16 = 32768
constexpr size_t WS_WFC1  = 131072;   // 128x512 bf16 = 131072
constexpr size_t WS_WFC2  = 262144;   // 512x128 bf16 = 131072
constexpr size_t WS_RELB  = 393216;   // 64x64 f32    = 16384

DEV v8f zero8() {
    v8f z;
#pragma unroll
    for (int i = 0; i < 8; ++i) z[i] = 0.0f;
    return z;
}

DEV v16bf pack16(bf16x8 lo, bf16x8 hi) {
    v16bf f;
#pragma unroll
    for (int i = 0; i < 8; ++i) { f[i] = lo[i]; f[i + 8] = hi[i]; }
    return f;
}

// A-fragment (16x32 bf16, M x K): lane l<16 -> row m0+l, K {k0..k0+15};
// lane>=16 same row group, K {k0+16..k0+31}.  Two b128 loads.
DEV v16bf fragA(const bf16_t* base, int pitch, int m0, int k0, int lane) {
    const bf16_t* p = base + (m0 + (lane & 15)) * pitch + k0 + ((lane >> 4) << 4);
    bf16x8 lo = *(const bf16x8*)p;
    bf16x8 hi = *(const bf16x8*)(p + 8);
    return pack16(lo, hi);
}

// B-fragment (32x16 bf16, K x N) from a [N][K]-major buffer (i.e. transposed source):
// lane holds column n0+(l&15); K = k0 + 16*(l>>4) + e (16 consecutive) -> one 32B read.
DEV v16bf fragB(const bf16_t* base, int pitch, int n0, int k0, int lane) {
    const bf16_t* p = base + (n0 + (lane & 15)) * pitch + k0 + ((lane >> 4) << 4);
    bf16x8 lo = *(const bf16x8*)p;
    bf16x8 hi = *(const bf16x8*)(p + 8);
    return pack16(lo, hi);
}

// B-fragment from pre-swizzled weight tiles in global: lane reads 32 contiguous bytes.
DEV v16bf fragW(const bf16_t* w, int ntiles, int kt, int nt, int lane) {
    const bf16_t* p = w + (((size_t)(kt * ntiles + nt)) * 32 + lane) * 16;
    bf16x8 lo = *(const bf16x8*)p;
    bf16x8 hi = *(const bf16x8*)(p + 8);
    return pack16(lo, hi);
}

DEV v8f wmma_bf16(v16bf a, v16bf b, v8f c) {
    return __builtin_amdgcn_wmma_f32_16x16x32_bf16(false, a, false, b, (short)0, c, false, false);
}

DEV float redsum16(float v) {
#pragma unroll
    for (int m = 1; m < 16; m <<= 1) v += __shfl_xor(v, m, 32);
    return v;
}
DEV float redmax16(float v) {
#pragma unroll
    for (int m = 1; m < 16; m <<= 1) v = fmaxf(v, __shfl_xor(v, m, 32));
    return v;
}

// ---- prep: swizzle f32 weights [K][N] into bf16 B-fragment tile order ----
__global__ void prep_weight(const float* __restrict__ W, bf16_t* __restrict__ out,
                            int K, int N) {
    int tid = blockIdx.x * blockDim.x + threadIdx.x;
    if (tid >= K * N) return;
    int e    = tid & 15;
    int lane = (tid >> 4) & 31;
    int tile = tid >> 9;
    int ntiles = N >> 4;
    int kt = tile / ntiles, nt = tile % ntiles;
    int k = kt * 32 + ((lane >> 4) << 4) + e;
    int n = nt * 16 + (lane & 15);
    out[tid] = (bf16_t)W[(size_t)k * N + n];
}

__global__ void prep_relb(const float* __restrict__ bias_values,
                          const int* __restrict__ pos_idx, float* __restrict__ relb) {
    int i = blockIdx.x * blockDim.x + threadIdx.x;
    if (i < Lx * Lx) relb[i] = bias_values[pos_idx[i]];
}

// ---- fused transformer block: one workgroup per batch element ----
__global__ void __launch_bounds__(128, 1)
block_kernel(const float* __restrict__ x, const float* __restrict__ gamma,
             const float* __restrict__ beta,
             const bf16_t* __restrict__ wqkv, const float* __restrict__ bqkv,
             const float* __restrict__ relb,
             const bf16_t* __restrict__ wproj, const float* __restrict__ bproj,
             const bf16_t* __restrict__ wfc1, const float* __restrict__ bfc1,
             const bf16_t* __restrict__ wfc2, const float* __restrict__ bfc2,
             float* __restrict__ out) {
    __shared__ __align__(16) char smem[65536];
    bf16_t* qk  = (bf16_t*)smem;             // [64][256]  Q cols 0..127 | K cols 128..255
    bf16_t* vt  = (bf16_t*)(smem + 32768);   // [128][64]  V transposed; later hh [64][128]
    bf16_t* nxr = (bf16_t*)(smem + 49152);   // [64][128]  nx / P-scratch / n1
    float*  xs  = (float*)smem;              // [64][129]  phase-1 x staging

    const int tid  = threadIdx.x;
    const int lane = tid & 31;
    const int wv   = tid >> 5;    // wave id 0..3  (= m-tile, = head)
    const int l15  = lane & 15;
    const int half = lane >> 4;
    const size_t xbase = (size_t)blockIdx.x * (Lx * Cc);

    // ---------- Phase 1: load x, LayerNorm -> nxr (bf16) ----------
    for (int i = tid; i < Lx * Cc; i += 128) {
        int r = i >> 7, c = i & 127;
        xs[r * 129 + c] = x[xbase + i];
    }
    __syncthreads();
    if (tid < Lx) {
        const float* row = xs + tid * 129;
        float s = 0.0f;
        for (int c = 0; c < Cc; ++c) s += row[c];
        float mu = s * (1.0f / Cc);
        float v = 0.0f;
        for (int c = 0; c < Cc; ++c) { float d = row[c] - mu; v += d * d; }
        float rs = rsqrtf(v * (1.0f / Cc) + 1e-5f);
        for (int c = 0; c < Cc; ++c)
            nxr[tid * Cc + c] = (bf16_t)((row[c] - mu) * rs * gamma[c] + beta[c]);
    }
    __syncthreads();

    // ---------- Phase 2: QKV = nx @ Wqkv (+bias); V stored transposed ----------
    {
        const int mt = wv;
        v16bf a[4];
#pragma unroll
        for (int kt = 0; kt < 4; ++kt) a[kt] = fragA(nxr, Cc, mt * 16, kt * 32, lane);
        for (int nt = 0; nt < 24; ++nt) {
            v8f acc = zero8();
#pragma unroll
            for (int kt = 0; kt < 4; ++kt)
                acc = wmma_bf16(a[kt], fragW(wqkv, 24, kt, nt, lane), acc);
            const int col = nt * 16 + l15;
            const float bq = bqkv[col];
            if (nt < 16) {
#pragma unroll
                for (int r = 0; r < 8; ++r) {
                    int row = mt * 16 + r + 8 * half;
                    qk[row * 256 + col] = (bf16_t)(acc[r] + bq);
                }
            } else {
#pragma unroll
                for (int r = 0; r < 8; ++r) {
                    int row = mt * 16 + r + 8 * half;
                    vt[(col - 256) * 64 + row] = (bf16_t)(acc[r] + bq);
                }
            }
        }
    }
    __syncthreads();

    // ---------- Phase 3: attention, head h = wave ----------
    {
        const int h = wv;
        bf16_t* ps = nxr + wv * 1024;   // per-wave P scratch [16][64]
        const float sc = 0.17677669529663689f;  // 1/sqrt(32)
        for (int mt = 0; mt < 4; ++mt) {
            v16bf aq = fragA(qk, 256, mt * 16, h * 32, lane);
            v8f S[4];
#pragma unroll
            for (int nt = 0; nt < 4; ++nt) {
                v16bf bk = fragB(qk, 256, nt * 16, 128 + h * 32, lane);  // K rows = B
                S[nt] = wmma_bf16(aq, bk, zero8());
            }
#pragma unroll
            for (int nt = 0; nt < 4; ++nt)
#pragma unroll
                for (int r = 0; r < 8; ++r) {
                    int row = mt * 16 + r + 8 * half;
                    S[nt][r] = S[nt][r] * sc + relb[row * 64 + nt * 16 + l15];
                }
            // softmax over each row (64 cols live in 16-lane half x 4 tiles)
#pragma unroll
            for (int r = 0; r < 8; ++r) {
                float m = S[0][r];
#pragma unroll
                for (int nt = 1; nt < 4; ++nt) m = fmaxf(m, S[nt][r]);
                m = redmax16(m);
                float e[4], s = 0.0f;
#pragma unroll
                for (int nt = 0; nt < 4; ++nt) { e[nt] = __expf(S[nt][r] - m); s += e[nt]; }
                s = redsum16(s);
                float inv = 1.0f / s;
                int prow = r + 8 * half;
#pragma unroll
                for (int nt = 0; nt < 4; ++nt)
                    ps[prow * 64 + nt * 16 + l15] = (bf16_t)(e[nt] * inv);
            }
            // O = P @ V  (vt is [d][token] so it serves B directly)
            v8f O0 = zero8(), O1 = zero8();
#pragma unroll
            for (int kt = 0; kt < 2; ++kt) {
                v16bf ap = fragA(ps, 64, 0, kt * 32, lane);
                O0 = wmma_bf16(ap, fragB(vt, 64, h * 32, kt * 32, lane), O0);
                O1 = wmma_bf16(ap, fragB(vt, 64, h * 32 + 16, kt * 32, lane), O1);
            }
            // O overwrites the (dead) Q columns of this head
#pragma unroll
            for (int r = 0; r < 8; ++r) {
                int row = mt * 16 + r + 8 * half;
                qk[row * 256 + h * 32 + l15]      = (bf16_t)O0[r];
                qk[row * 256 + h * 32 + 16 + l15] = (bf16_t)O1[r];
            }
        }
    }
    __syncthreads();

    // ---------- Phase 4: out1 = x + O @ Wproj + b; LN2 -> nxr ----------
    v8f o1[8];
    {
        const int mt = wv;
        v16bf a[4];
#pragma unroll
        for (int kt = 0; kt < 4; ++kt) a[kt] = fragA(qk, 256, mt * 16, kt * 32, lane);
#pragma unroll
        for (int nt = 0; nt < 8; ++nt) {
            v8f acc = zero8();
#pragma unroll
            for (int kt = 0; kt < 4; ++kt)
                acc = wmma_bf16(a[kt], fragW(wproj, 8, kt, nt, lane), acc);
            int col = nt * 16 + l15;
            float bp = bproj[col];
#pragma unroll
            for (int r = 0; r < 8; ++r) {
                int row = mt * 16 + r + 8 * half;
                acc[r] += bp + x[xbase + row * 128 + col];
            }
            o1[nt] = acc;
        }
        float gcol[8], bcol[8];
#pragma unroll
        for (int nt = 0; nt < 8; ++nt) { int c = nt * 16 + l15; gcol[nt] = gamma[c]; bcol[nt] = beta[c]; }
#pragma unroll
        for (int r = 0; r < 8; ++r) {
            float s = 0.0f;
#pragma unroll
            for (int nt = 0; nt < 8; ++nt) s += o1[nt][r];
            s = redsum16(s);
            float mu = s * (1.0f / 128.0f);
            float v = 0.0f;
#pragma unroll
            for (int nt = 0; nt < 8; ++nt) { float d = o1[nt][r] - mu; v += d * d; }
            v = redsum16(v);
            float rs = rsqrtf(v * (1.0f / 128.0f) + 1e-5f);
            int row = mt * 16 + r + 8 * half;
#pragma unroll
            for (int nt = 0; nt < 8; ++nt) {
                int col = nt * 16 + l15;
                nxr[row * 128 + col] = (bf16_t)((o1[nt][r] - mu) * rs * gcol[nt] + bcol[nt]);
            }
        }
    }
    // no barrier needed: each wave only touches its own 16 rows from here on

    // ---------- Phase 5: MLP (4 chunks of 128 cols), fused GELU, final residual ----------
    {
        const int mt = wv;
        bf16_t* hh = vt;   // reuse dead V^T region as [64][128]
        v16bf a1[4];
#pragma unroll
        for (int kt = 0; kt < 4; ++kt) a1[kt] = fragA(nxr, 128, mt * 16, kt * 32, lane);
        v8f macc[8];
#pragma unroll
        for (int nt = 0; nt < 8; ++nt) macc[nt] = zero8();
        for (int cc = 0; cc < 4; ++cc) {
#pragma unroll
            for (int nt = 0; nt < 8; ++nt) {
                v8f acc = zero8();
#pragma unroll
                for (int kt = 0; kt < 4; ++kt)
                    acc = wmma_bf16(a1[kt], fragW(wfc1, 32, kt, cc * 8 + nt, lane), acc);
                int col = nt * 16 + l15;
                float b1 = bfc1[cc * 128 + col];
#pragma unroll
                for (int r = 0; r < 8; ++r) {
                    float u = acc[r] + b1;
                    float g = 0.5f * u * (1.0f + erff(u * 0.70710678118f));
                    int row = mt * 16 + r + 8 * half;
                    hh[row * 128 + col] = (bf16_t)g;
                }
            }
            v16bf a2[4];
#pragma unroll
            for (int kk = 0; kk < 4; ++kk) a2[kk] = fragA(hh, 128, mt * 16, kk * 32, lane);
#pragma unroll
            for (int nt = 0; nt < 8; ++nt)
#pragma unroll
                for (int kk = 0; kk < 4; ++kk)
                    macc[nt] = wmma_bf16(a2[kk], fragW(wfc2, 8, cc * 4 + kk, nt, lane), macc[nt]);
        }
#pragma unroll
        for (int nt = 0; nt < 8; ++nt) {
            int col = nt * 16 + l15;
            float b2 = bfc2[col];
#pragma unroll
            for (int r = 0; r < 8; ++r) {
                int row = mt * 16 + r + 8 * half;
                out[xbase + row * 128 + col] = o1[nt][r] + macc[nt][r] + b2;
            }
        }
    }
}

extern "C" void kernel_launch(void* const* d_in, const int* in_sizes, int n_in,
                              void* d_out, int out_size, void* d_ws, size_t ws_size,
                              hipStream_t stream) {
    (void)in_sizes; (void)n_in; (void)out_size; (void)ws_size;
    const float* x           = (const float*)d_in[0];
    const float* gamma       = (const float*)d_in[1];
    const float* beta        = (const float*)d_in[2];
    const float* w_qkv       = (const float*)d_in[3];
    const float* b_qkv       = (const float*)d_in[4];
    const float* bias_values = (const float*)d_in[5];
    const float* w_proj      = (const float*)d_in[6];
    const float* b_proj      = (const float*)d_in[7];
    const float* w_fc1       = (const float*)d_in[8];
    const float* b_fc1       = (const float*)d_in[9];
    const float* w_fc2       = (const float*)d_in[10];
    const float* b_fc2       = (const float*)d_in[11];
    const int*   pos_idx     = (const int*)d_in[12];

    char* ws = (char*)d_ws;
    bf16_t* wqkv  = (bf16_t*)(ws + WS_WQKV);
    bf16_t* wproj = (bf16_t*)(ws + WS_WPROJ);
    bf16_t* wfc1  = (bf16_t*)(ws + WS_WFC1);
    bf16_t* wfc2  = (bf16_t*)(ws + WS_WFC2);
    float*  relb  = (float*)(ws + WS_RELB);

    prep_weight<<<(128 * 384 + 255) / 256, 256, 0, stream>>>(w_qkv, wqkv, 128, 384);
    prep_weight<<<(128 * 128 + 255) / 256, 256, 0, stream>>>(w_proj, wproj, 128, 128);
    prep_weight<<<(128 * 512 + 255) / 256, 256, 0, stream>>>(w_fc1, wfc1, 128, 512);
    prep_weight<<<(512 * 128 + 255) / 256, 256, 0, stream>>>(w_fc2, wfc2, 512, 128);
    prep_relb<<<(Lx * Lx + 255) / 256, 256, 0, stream>>>(bias_values, pos_idx, relb);

    block_kernel<<<Bb, 128, 0, stream>>>(x, gamma, beta, wqkv, b_qkv, relb,
                                         wproj, b_proj, wfc1, b_fc1, wfc2, b_fc2,
                                         (float*)d_out);
}
